// GeometryEncoder_25984552140950
// MI455X (gfx1250) — matrix operand order repaired and checked by reading, not compiled, not run on hardware
//
#include <hip/hip_runtime.h>
#include <hip/hip_bf16.h>
#include <cstdint>
#include <cstddef>

#define N_NODES 16384
#define T_TRI   65536
#define H_DIM   128
#define G_POOL  512
#define NIT     4
#define NL      4
#define INV_F   0.9999950000374996f
#define LSTRIDE 136    // bf16 elems per LDS row (128 + 8 pad -> bank-conflict free)
#define NTILES  4104   // 513 blocks * 8 waves; >= worst-case padded tiles (4099)

typedef __bf16 bf16;
typedef __attribute__((ext_vector_type(16))) __bf16 v16bf;
typedef __attribute__((ext_vector_type(8)))  float  v8f;

// ---------------------------------------------------------------- utilities
__global__ void fill_val_u32(uint32_t* __restrict__ p, uint32_t v, int n) {
  int i = blockIdx.x * 256 + threadIdx.x;
  if (i < n) p[i] = v;
}

// Fold BatchNorm(eval) + bias:  y = z*s + c,  s = gamma*INV, c = b*s + beta
__global__ void scales_kernel(const float* __restrict__ b0, const float* __restrict__ g0,
                              const float* __restrict__ be0, const float* __restrict__ bh,
                              const float* __restrict__ gh, const float* __restrict__ beh,
                              float* __restrict__ s0, float* __restrict__ c0,
                              float* __restrict__ sh, float* __restrict__ ch) {
  int i = blockIdx.x * 256 + threadIdx.x;
  if (i < NIT * 4 * H_DIM)      { float s = g0[i] * INV_F; s0[i] = s; c0[i] = b0[i] * s + be0[i]; }
  if (i < NIT * NL * 4 * H_DIM) { float s = gh[i] * INV_F; sh[i] = s; ch[i] = bh[i] * s + beh[i]; }
}

// ------------------------------------------------- pack weights as B-fragments
// Per matrix: [k_tile][n_tile][lane(32)][elem(16)] bf16; elem e of lane l is
// W[kt*32 + (l>=16?16:0) + e][nt*16 + (l&15)]  (contiguous k = 16h + e, same order as the A-fragment loads).
__global__ void pack_w0_kernel(const float* __restrict__ W0, bf16* __restrict__ pW0) {
  int tid  = blockIdx.x * 256 + threadIdx.x;      // 16 * 16*8*32*16 = 1,048,576
  int e    = tid & 15;
  int lane = (tid >> 4) & 31;
  int nt   = (tid >> 9) & 7;
  int kt   = (tid >> 12) & 15;
  int mg   = tid >> 16;                           // it*4 + g
  int K    = kt * 32 + ((lane >> 4) << 4) + e;
  int Nc   = nt * 16 + (lane & 15);
  pW0[tid] = (bf16)W0[(size_t)mg * (512 * 128) + (size_t)K * 128 + Nc];
}

__global__ void pack_wh_kernel(const float* __restrict__ Wh, bf16* __restrict__ pWh) {
  int tid  = blockIdx.x * 256 + threadIdx.x;      // 64 * 4*8*32*16 = 1,048,576
  int e    = tid & 15;
  int lane = (tid >> 4) & 31;
  int nt   = (tid >> 9) & 7;
  int kt   = (tid >> 12) & 3;
  int m    = tid >> 14;                           // (it*NL + l)*4 + g
  int K    = kt * 32 + ((lane >> 4) << 4) + e;
  int Nc   = nt * 16 + (lane & 15);
  pWh[tid] = (bf16)Wh[(size_t)m * (128 * 128) + (size_t)K * 128 + Nc];
}

// ------------------------------------------------------- geometry encoder
__global__ void geo_kernel(const float* __restrict__ pos, const int* __restrict__ EI,
                           const int* __restrict__ ejk, const int* __restrict__ eij,
                           const int* __restrict__ nei_p,
                           const float* __restrict__ Wg, const float* __restrict__ bg,
                           const float* __restrict__ gamg, const float* __restrict__ betg,
                           bf16* __restrict__ geo, unsigned char* __restrict__ grp) {
  int t = blockIdx.x;
  int h = threadIdx.x;
  int i = EI[t], j = EI[T_TRI + t], k = EI[2 * T_TRI + t];
  float v1x = pos[2 * j] - pos[2 * i],     v1y = pos[2 * j + 1] - pos[2 * i + 1];
  float v2x = pos[2 * k] - pos[2 * j],     v2y = pos[2 * k + 1] - pos[2 * j + 1];
  float d1  = sqrtf(v1x * v1x + v1y * v1y);
  float d2  = sqrtf(v2x * v2x + v2y * v2y);
  float cr  = v1x * v2y - v1y * v2x;
  float dt  = v1x * v2x + v1y * v2y;
  float th  = atan2f(fabsf(cr), dt);
  float y = d1 * Wg[h] + d2 * Wg[H_DIM + h] + th * Wg[2 * H_DIM + h] + bg[h];
  y = y * (gamg[h] * INV_F) + betg[h];
  y = fmaxf(y, 0.0f);
  geo[(size_t)t * H_DIM + h] = (bf16)y;
  if (h == 0) {
    int nei = nei_p[0];
    int in1 = eij[t] < nei;
    int in2 = ejk[t] < nei;
    grp[t] = (unsigned char)((in1 ? 0 : 2) + (in2 ? 0 : 1));  // matches mask stack order
  }
}

// ------------------------------------- deterministic stable partition by group
__global__ void hist_kernel(const unsigned char* __restrict__ grp, int* __restrict__ hist) {
  __shared__ int cnt[4];
  if (threadIdx.x < 4) cnt[threadIdx.x] = 0;
  __syncthreads();
  atomicAdd(&cnt[grp[blockIdx.x * 256 + threadIdx.x]], 1);
  __syncthreads();
  if (threadIdx.x < 4) hist[blockIdx.x * 4 + threadIdx.x] = cnt[threadIdx.x];
}

__global__ void scan_kernel(const int* __restrict__ hist, int* __restrict__ bstart,
                            unsigned char* __restrict__ tileg) {
  if (threadIdx.x != 0 || blockIdx.x != 0) return;
  int total[4] = {0, 0, 0, 0};
  for (int b = 0; b < 256; ++b)
    for (int g = 0; g < 4; ++g) total[g] += hist[b * 4 + g];
  int base[5]; base[0] = 0;
  for (int g = 0; g < 4; ++g) base[g + 1] = base[g] + ((total[g] + 15) & ~15);
  int run[4] = {0, 0, 0, 0};
  for (int b = 0; b < 256; ++b)
    for (int g = 0; g < 4; ++g) { bstart[b * 4 + g] = base[g] + run[g]; run[g] += hist[b * 4 + g]; }
  for (int g = 0; g < 4; ++g)
    for (int t = base[g] >> 4; t < (base[g + 1] >> 4); ++t) tileg[t] = (unsigned char)g;
}

__global__ void scatter_kernel(const unsigned char* __restrict__ grp,
                               const int* __restrict__ bstart, int* __restrict__ perm) {
  __shared__ unsigned char sg[256];
  int b = blockIdx.x, i = threadIdx.x;
  sg[i] = grp[b * 256 + i];
  __syncthreads();
  int g = sg[i];
  int rank = 0;
  for (int j = 0; j < i; ++j) rank += (sg[j] == g) ? 1 : 0;   // stable within block
  perm[bstart[b * 4 + g] + rank] = b * 256 + i;
}

// --------------------------------------------------------------- main MLP
// One wave per group-uniform 16-triplet tile (triplets permuted by group, pad
// rows have perm = -1). Layer0 (K=512) + NL hidden (K=128) via
// v_wmma_f32_16x16x32_bf16; B fragments preloaded 4 at a time so global b128
// loads overlap the matrix pipe. Activations bounce through per-wave LDS.
__global__ void __launch_bounds__(256)
mlp_kernel(const bf16* __restrict__ nf_in, const bf16* __restrict__ geo,
           const int* __restrict__ perm, const unsigned char* __restrict__ tileg,
           const int* __restrict__ EI,
           const bf16* __restrict__ pW0, const bf16* __restrict__ pWh,
           const float* __restrict__ s0, const float* __restrict__ c0,
           const float* __restrict__ sh, const float* __restrict__ ch,
           const float* __restrict__ att, const bf16* __restrict__ zrow,
           float* __restrict__ nf_out, int it) {
  __shared__ __align__(16) bf16 hbuf[8][16 * LSTRIDE];

  const int lane = threadIdx.x & 31;
  const int wv   = threadIdx.x >> 5;
  const int tile = blockIdx.x * 8 + wv;
  const int g    = __builtin_amdgcn_readfirstlane((int)tileg[tile]);
  if (g > 3) return;                                // unused padding tile (uniform per wave)

  const int half = lane >> 4;
  const int colb = lane & 15;
  const int kA   = half * 16;                       // A-fragment K offset (contiguous k = 16h + s)

  // Layer-0 A-matrix row gathers (row = lane&15 of the tile; pad rows -> zeros).
  const int trowA = perm[tile * 16 + colb];
  const bf16 *seg0, *seg1, *seg2, *seg3;
  if (trowA >= 0) {
    seg0 = nf_in + (size_t)EI[trowA] * H_DIM;
    seg1 = nf_in + (size_t)EI[T_TRI + trowA] * H_DIM;
    seg2 = nf_in + (size_t)EI[2 * T_TRI + trowA] * H_DIM;
    seg3 = geo + (size_t)trowA * H_DIM;
  } else {
    seg0 = seg1 = seg2 = seg3 = zrow;
  }

  // C/D-layout rows handled by this lane: m = half*8 + r
  int tC[8], irow[8];
#pragma unroll
  for (int r = 0; r < 8; ++r) {
    tC[r] = perm[tile * 16 + half * 8 + r];
    irow[r] = (tC[r] >= 0) ? EI[tC[r]] : 0;
  }

  bf16* lbuf = hbuf[wv];

  // ---------------- layer 0: [16 x 512] @ [512 x 128] ----------------
  const bf16* w0p = pW0 + (size_t)(it * 4 + g) * (16 * 8 * 512);
  v8f acc[8];
#pragma unroll
  for (int nt = 0; nt < 8; ++nt)
#pragma unroll
    for (int q = 0; q < 8; ++q) acc[nt][q] = 0.0f;

#pragma unroll
  for (int kt = 0; kt < 16; ++kt) {
    const bf16* sp = (kt < 4) ? seg0 : (kt < 8) ? seg1 : (kt < 12) ? seg2 : seg3;
    const bf16* ap = sp + (kt & 3) * 32 + kA;
    v16bf afr;
    ((uint4*)&afr)[0] = *(const uint4*)(ap);
    ((uint4*)&afr)[1] = *(const uint4*)(ap + 8);
    const bf16* wp = w0p + ((size_t)kt * 8 * 32 + lane) * 16;
#pragma unroll
    for (int nh = 0; nh < 2; ++nh) {                 // 4 B-fragments per batch
      v16bf bfr[4];
#pragma unroll
      for (int nn = 0; nn < 4; ++nn) {
        const bf16* bp = wp + (size_t)(nh * 4 + nn) * 512;
        ((uint4*)&bfr[nn])[0] = *(const uint4*)(bp);
        ((uint4*)&bfr[nn])[1] = *(const uint4*)(bp + 8);
      }
#pragma unroll
      for (int nn = 0; nn < 4; ++nn)
        acc[nh * 4 + nn] = __builtin_amdgcn_wmma_f32_16x16x32_bf16(
            false, afr, false, bfr[nn], (short)0, acc[nh * 4 + nn], false, false);
    }
  }
  {
    const float* sp = s0 + (size_t)(it * 4 + g) * H_DIM;
    const float* cp = c0 + (size_t)(it * 4 + g) * H_DIM;
#pragma unroll
    for (int nt = 0; nt < 8; ++nt) {
      const int c = nt * 16 + colb;
      const float s = sp[c], b = cp[c];
#pragma unroll
      for (int r = 0; r < 8; ++r) {
        float y = fmaxf(acc[nt][r] * s + b, 0.0f);
        lbuf[(half * 8 + r) * LSTRIDE + c] = (bf16)y;
      }
    }
  }

  // ---------------- hidden layers: [16 x 128] @ [128 x 128] ----------------
  for (int l = 0; l < NL; ++l) {
    const int m = (it * NL + l) * 4 + g;
    const bf16* whp = pWh + (size_t)m * (4 * 8 * 512);
    v8f acc2[8];
#pragma unroll
    for (int nt = 0; nt < 8; ++nt)
#pragma unroll
      for (int q = 0; q < 8; ++q) acc2[nt][q] = 0.0f;

#pragma unroll
    for (int kt = 0; kt < 4; ++kt) {
      const bf16* ap = lbuf + colb * LSTRIDE + kt * 32 + kA;
      v16bf afr;
      ((uint4*)&afr)[0] = *(const uint4*)(ap);
      ((uint4*)&afr)[1] = *(const uint4*)(ap + 8);
      const bf16* wp = whp + ((size_t)kt * 8 * 32 + lane) * 16;
#pragma unroll
      for (int nh = 0; nh < 2; ++nh) {
        v16bf bfr[4];
#pragma unroll
        for (int nn = 0; nn < 4; ++nn) {
          const bf16* bp = wp + (size_t)(nh * 4 + nn) * 512;
          ((uint4*)&bfr[nn])[0] = *(const uint4*)(bp);
          ((uint4*)&bfr[nn])[1] = *(const uint4*)(bp + 8);
        }
#pragma unroll
        for (int nn = 0; nn < 4; ++nn)
          acc2[nh * 4 + nn] = __builtin_amdgcn_wmma_f32_16x16x32_bf16(
              false, afr, false, bfr[nn], (short)0, acc2[nh * 4 + nn], false, false);
      }
    }
    const float* sp = sh + (size_t)m * H_DIM;
    const float* cp = ch + (size_t)m * H_DIM;
    if (l < NL - 1) {
#pragma unroll
      for (int nt = 0; nt < 8; ++nt) {
        const int c = nt * 16 + colb;
        const float s = sp[c], b = cp[c];
#pragma unroll
        for (int r = 0; r < 8; ++r) {
          float y = fmaxf(acc2[nt][r] * s + b, 0.0f);
          lbuf[(half * 8 + r) * LSTRIDE + c] = (bf16)y;
        }
      }
    } else {
      // relu output >= 0 -> leaky_relu is identity; scale by att[g]; scatter-add.
      const float ag = att[g];
#pragma unroll
      for (int r = 0; r < 8; ++r) {
        if (tC[r] >= 0) {
          const size_t rb = (size_t)irow[r] * H_DIM;
#pragma unroll
          for (int nt = 0; nt < 8; ++nt) {
            const int c = nt * 16 + colb;
            float y = fmaxf(acc2[nt][r] * sp[c] + cp[c], 0.0f) * ag;
            unsafeAtomicAdd(nf_out + rb + c, y);
          }
        }
      }
    }
  }
}

// ------------------------------------------- per-interaction epilogue + pool
__global__ void finish_kernel(const float* __restrict__ nf, bf16* __restrict__ nfb,
                              const int* __restrict__ ntf, float* __restrict__ out, int it) {
  int idx = blockIdx.x * 256 + threadIdx.x;       // N * 128
  float v = nf[idx];
  nfb[idx] = (bf16)v;                             // bf16 copy for next interaction's gathers
  int n = idx >> 7, c = idx & 127;
  unsafeAtomicAdd(out + ((size_t)ntf[n] * (NIT * H_DIM) + it * H_DIM + c), v);
}

__global__ void clamp_kernel(float* __restrict__ out) {
  int i = blockIdx.x * 256 + threadIdx.x;
  if (i < G_POOL * NIT * H_DIM) out[i] = fminf(out[i], 1.0e6f);
}

// ---------------------------------------------------------------- launcher
extern "C" void kernel_launch(void* const* d_in, const int* in_sizes, int n_in,
                              void* d_out, int out_size, void* d_ws, size_t ws_size,
                              hipStream_t stream) {
  (void)in_sizes; (void)n_in; (void)out_size; (void)ws_size;
  const float* pos  = (const float*)d_in[0];
  const int*   EI   = (const int*)d_in[1];
  const int*   ejk  = (const int*)d_in[2];
  const int*   eij  = (const int*)d_in[3];
  const int*   nei  = (const int*)d_in[4];
  const int*   ntf  = (const int*)d_in[5];
  const float* Wg   = (const float*)d_in[6];
  const float* bg   = (const float*)d_in[7];
  const float* gamg = (const float*)d_in[8];
  const float* betg = (const float*)d_in[9];
  const float* W0   = (const float*)d_in[10];
  const float* b0   = (const float*)d_in[11];
  const float* g0   = (const float*)d_in[12];
  const float* be0  = (const float*)d_in[13];
  const float* Wh   = (const float*)d_in[14];
  const float* bh   = (const float*)d_in[15];
  const float* gh   = (const float*)d_in[16];
  const float* beh  = (const float*)d_in[17];
  const float* att  = (const float*)d_in[18];
  float* out = (float*)d_out;

  char* ws = (char*)d_ws;
  bf16*          geo   = (bf16*)(ws);                     // T*128 bf16   = 16,777,216 B
  bf16*          nfb   = (bf16*)(ws + 16777216);          // N*128 bf16   =  4,194,304 B
  float*         nfn   = (float*)(ws + 20971520);         // N*128 f32    =  8,388,608 B
  bf16*          pw0   = (bf16*)(ws + 29360128);          // packed W0    =  2,097,152 B
  bf16*          pwh   = (bf16*)(ws + 31457280);          // packed Wh    =  2,097,152 B
  float*         s0    = (float*)(ws + 33554432);
  float*         c0    = (float*)(ws + 33562624);
  float*         sh    = (float*)(ws + 33570816);
  float*         ch    = (float*)(ws + 33603584);
  unsigned char* grp   = (unsigned char*)(ws + 33636352); // T bytes
  int*           hist  = (int*)(ws + 33701888);           // 256*4 ints
  int*           bst   = (int*)(ws + 33705984);           // 256*4 ints
  int*           perm  = (int*)(ws + 33710080);           // NTILES*16 ints = 262,656 B
  unsigned char* tilg  = (unsigned char*)(ws + 33972736); // NTILES bytes (4104)
  bf16*          zrow  = (bf16*)(ws + 33976848);          // 128 bf16 zeros

  fill_val_u32<<<262144 / 256, 256, 0, stream>>>((uint32_t*)out, 0u, 262144);
  fill_val_u32<<<1048576 / 256, 256, 0, stream>>>((uint32_t*)nfb, 0u, 1048576);
  fill_val_u32<<<1, 64, 0, stream>>>((uint32_t*)zrow, 0u, 64);
  scales_kernel<<<8192 / 256, 256, 0, stream>>>(b0, g0, be0, bh, gh, beh, s0, c0, sh, ch);
  pack_w0_kernel<<<1048576 / 256, 256, 0, stream>>>(W0, pw0);
  pack_wh_kernel<<<1048576 / 256, 256, 0, stream>>>(Wh, pwh);
  geo_kernel<<<T_TRI, 128, 0, stream>>>(pos, EI, ejk, eij, nei, Wg, bg, gamg, betg, geo, grp);

  // deterministic stable partition of triplets by edge-type group
  fill_val_u32<<<(NTILES * 16 + 255) / 256, 256, 0, stream>>>((uint32_t*)perm, 0xFFFFFFFFu,
                                                              NTILES * 16);
  fill_val_u32<<<(NTILES / 4 + 255) / 256, 256, 0, stream>>>((uint32_t*)tilg, 0xFFFFFFFFu,
                                                             NTILES / 4);
  hist_kernel<<<256, 256, 0, stream>>>(grp, hist);
  scan_kernel<<<1, 1, 0, stream>>>(hist, bst, tilg);
  scatter_kernel<<<256, 256, 0, stream>>>(grp, bst, perm);

  for (int it = 0; it < NIT; ++it) {
    fill_val_u32<<<2097152 / 256, 256, 0, stream>>>((uint32_t*)nfn, 0u, 2097152);
    mlp_kernel<<<NTILES / 8, 256, 0, stream>>>(nfb, geo, perm, tilg, EI, pw0, pwh,
                                               s0, c0, sh, ch, att, zrow, nfn, it);
    finish_kernel<<<2097152 / 256, 256, 0, stream>>>(nfn, nfb, ntf, out, it);
  }
  clamp_kernel<<<262144 / 256, 256, 0, stream>>>(out);
}
